// MyLSTMCell_23398981829078
// MI455X (gfx1250) — matrix operand order repaired and checked by reading, not compiled, not run on hardware
//
#include <hip/hip_runtime.h>
#include <hip/hip_bf16.h>

typedef __attribute__((ext_vector_type(16))) _Float16 v16h;
typedef __attribute__((ext_vector_type(4)))  _Float16 v4h;
typedef __attribute__((ext_vector_type(8)))  float    v8f;
typedef __attribute__((ext_vector_type(4)))  int      v4i;

#define B_DIM  16384
#define IN_DIM 1024
#define H_DIM  1024
#define K_DIM  2048   // IN + H
#define BM 128
#define BN 32         // gate columns per block (x4 gates)
#define KT 64
#define LDT 72        // padded LDS stride (halfs): 144B, 16B-aligned
#define NTILE (K_DIM / KT)

#if defined(__gfx1250__) && __has_builtin(__builtin_amdgcn_global_load_async_to_lds_b128)
#define USE_ASYNC_LDS 1
#else
#define USE_ASYNC_LDS 0
#endif

// ---------------- pack kernels: f32 -> f16, K-concatenated ----------------

__global__ __launch_bounds__(256)
void pack_X_kernel(const float* __restrict__ input_,
                   const float* __restrict__ prev_h,
                   _Float16* __restrict__ X16) {
  int t  = blockIdx.x * blockDim.x + threadIdx.x;   // over B * (K/4)
  int m  = t >> 9;                                  // /512
  int kq = (t & 511) << 2;                          // *4
  const float* src = (kq < IN_DIM) ? (input_ + (size_t)m * IN_DIM + kq)
                                   : (prev_h + (size_t)m * H_DIM + (kq - IN_DIM));
  float4 v = *(const float4*)src;
  v4h h;
  h.x = (_Float16)v.x; h.y = (_Float16)v.y; h.z = (_Float16)v.z; h.w = (_Float16)v.w;
  *(v4h*)(X16 + (size_t)m * K_DIM + kq) = h;
}

__global__ __launch_bounds__(256)
void pack_W_kernel(const float* __restrict__ Wi0, const float* __restrict__ Wh0,
                   const float* __restrict__ Wi1, const float* __restrict__ Wh1,
                   const float* __restrict__ Wi2, const float* __restrict__ Wh2,
                   const float* __restrict__ Wi3, const float* __restrict__ Wh3,
                   _Float16* __restrict__ W16) {
  int t   = blockIdx.x * blockDim.x + threadIdx.x;  // over 4096 * 512
  int row = t >> 9;                                 // 0..4095
  int kq  = (t & 511) << 2;
  int g = row >> 10, j = row & 1023;
  const float* Wi = (g == 0) ? Wi0 : (g == 1) ? Wi1 : (g == 2) ? Wi2 : Wi3;
  const float* Wh = (g == 0) ? Wh0 : (g == 1) ? Wh1 : (g == 2) ? Wh2 : Wh3;
  const float* src = (kq < IN_DIM) ? (Wi + (size_t)j * IN_DIM + kq)
                                   : (Wh + (size_t)j * H_DIM + (kq - IN_DIM));
  float4 v = *(const float4*)src;
  v4h h;
  h.x = (_Float16)v.x; h.y = (_Float16)v.y; h.z = (_Float16)v.z; h.w = (_Float16)v.w;
  *(v4h*)(W16 + (size_t)row * K_DIM + kq) = h;
}

// ---------------- device helpers ----------------

__device__ __forceinline__ float sigmoidf_(float x) {
  return 1.0f / (1.0f + __expf(-x));
}

// tanh(x) = sign(x) * (1-e)/(1+e),  e = exp(-2|x|).  exp overflow -> +-1, monotone, smooth.
__device__ __forceinline__ float fast_tanhf_(float x) {
  float e = __expf(-2.0f * __builtin_fabsf(x));
  float t = (1.0f - e) / (1.0f + e);
  return __builtin_copysignf(t, x);
}

#if USE_ASYNC_LDS
typedef __attribute__((address_space(1))) v4i gv4i;   // global int4
typedef __attribute__((address_space(3))) v4i lv4i;   // LDS int4

__device__ __forceinline__ void wait_asynccnt_8() {
#if __has_builtin(__builtin_amdgcn_s_wait_asynccnt)
  __builtin_amdgcn_s_wait_asynccnt(8);
#else
  asm volatile("s_wait_asynccnt 0x8" ::: "memory");
#endif
}
__device__ __forceinline__ void wait_asynccnt_0() {
#if __has_builtin(__builtin_amdgcn_s_wait_asynccnt)
  __builtin_amdgcn_s_wait_asynccnt(0);
#else
  asm volatile("s_wait_asynccnt 0x0" ::: "memory");
#endif
}

// Issue one K-tile of async global->LDS DMA (8 x b128 per thread: 4 for A, 4 for B).
__device__ __forceinline__ void issue_tile_async(const _Float16* __restrict__ X16,
                                                 const _Float16* __restrict__ W16,
                                                 _Float16* Asb, _Float16* Bsb,
                                                 int bm, int bn, int kt, int tid) {
#pragma unroll
  for (int i = 0; i < 4; ++i) {
    int c = tid + 256 * i, row = c >> 3, kc = (c & 7) << 3;
    const _Float16* g = X16 + (size_t)(bm + row) * K_DIM + kt + kc;
    __builtin_amdgcn_global_load_async_to_lds_b128(
        (gv4i*)g, (lv4i*)(Asb + row * LDT + kc), 0, 0);
  }
#pragma unroll
  for (int i = 0; i < 4; ++i) {
    int c = tid + 256 * i, row = c >> 3, kc = (c & 7) << 3;
    int gt = row >> 5, col = row & 31;
    const _Float16* g = W16 + (size_t)(gt * H_DIM + bn + col) * K_DIM + kt + kc;
    __builtin_amdgcn_global_load_async_to_lds_b128(
        (gv4i*)g, (lv4i*)(Bsb + row * LDT + kc), 0, 0);
  }
}
#endif  // USE_ASYNC_LDS

// ---------------- fused LSTM GEMM + gate epilogue ----------------
// Block computes rows [bm,bm+128) x gate-cols [bn,bn+32) for ALL 4 gates,
// then applies sigmoid/tanh and writes h (out[0..B*H)) and c (out[B*H..2BH)).

__global__ __launch_bounds__(256)
void lstm_fused_kernel(const _Float16* __restrict__ X16,
                       const _Float16* __restrict__ W16,
                       const float* __restrict__ prev_c,
                       const float* __restrict__ b_ii, const float* __restrict__ b_hi,
                       const float* __restrict__ b_if, const float* __restrict__ b_hf,
                       const float* __restrict__ b_ig, const float* __restrict__ b_hg,
                       const float* __restrict__ b_io, const float* __restrict__ b_ho,
                       float* __restrict__ out) {
#if USE_ASYNC_LDS
  __shared__ _Float16 As[2][BM * LDT];        // 2 x 18 KB
  __shared__ _Float16 Bs[2][4 * BN * LDT];    // 2 x 18 KB
#else
  __shared__ _Float16 As[1][BM * LDT];
  __shared__ _Float16 Bs[1][4 * BN * LDT];
#endif

  const int tid  = threadIdx.x;
  const int lane = tid & 31;
  const int wave = tid >> 5;   // 0..7
  const int wm   = wave & 3;   // row group: 32 rows each
  const int wn   = wave >> 2;  // col group: 16 cols each
  const int lm   = lane & 15;
  const int lhi  = lane >> 4;

  const int bm = blockIdx.y * BM;
  const int bn = blockIdx.x * BN;

  v8f zero = {};
  v8f acc[4][2];
  #pragma unroll
  for (int g = 0; g < 4; ++g)
    #pragma unroll
    for (int mt = 0; mt < 2; ++mt) acc[g][mt] = zero;

#if USE_ASYNC_LDS
  issue_tile_async(X16, W16, As[0], Bs[0], bm, bn, 0, tid);
#endif

  for (int t = 0; t < NTILE; ++t) {
#if USE_ASYNC_LDS
    const int cur = t & 1;
    if (t + 1 < NTILE) {
      issue_tile_async(X16, W16, As[cur ^ 1], Bs[cur ^ 1], bm, bn, (t + 1) * KT, tid);
      wait_asynccnt_8();   // in-order completion: tile t fully resident
    } else {
      wait_asynccnt_0();
    }
    __syncthreads();       // all waves' DMAs for tile t visible
    const _Float16* Asb = As[cur];
    const _Float16* Bsb = Bs[cur];
#else
    __syncthreads();
    {
      // batched staging: all 8 global loads in flight before any LDS store
      float4 ra[4], rb[4];
      #pragma unroll
      for (int i = 0; i < 4; ++i) {
        int c = tid + 256 * i, row = c >> 3, kc = (c & 7) << 3;
        ra[i] = *(const float4*)(X16 + (size_t)(bm + row) * K_DIM + t * KT + kc);
        int gt = row >> 5, col = row & 31;
        rb[i] = *(const float4*)(W16 + (size_t)(gt * H_DIM + bn + col) * K_DIM + t * KT + kc);
      }
      #pragma unroll
      for (int i = 0; i < 4; ++i) {
        int c = tid + 256 * i, row = c >> 3, kc = (c & 7) << 3;
        *(float4*)(As[0] + row * LDT + kc) = ra[i];
        *(float4*)(Bs[0] + row * LDT + kc) = rb[i];
      }
    }
    __syncthreads();
    const _Float16* Asb = As[0];
    const _Float16* Bsb = Bs[0];
#endif

    #pragma unroll
    for (int ks = 0; ks < KT; ks += 32) {
      union { v16h v; float4 q[2]; } afr[2], bfr[4];
      // A fragments (16x32 f16): lane holds row M=lm; k-chunks at lhi*8 and 16+lhi*8
      #pragma unroll
      for (int mt = 0; mt < 2; ++mt) {
        int row = wm * 32 + mt * 16 + lm;
        afr[mt].q[0] = *(const float4*)(Asb + row * LDT + ks + lhi * 8);
        afr[mt].q[1] = *(const float4*)(Asb + row * LDT + ks + 16 + lhi * 8);
      }
      // B fragments (32x16 f16): lane holds col N=lm; k-chunks at lhi*8 and 16+lhi*8
      #pragma unroll
      for (int g = 0; g < 4; ++g) {
        int row = g * 32 + wn * 16 + lm;
        bfr[g].q[0] = *(const float4*)(Bsb + row * LDT + ks + lhi * 8);
        bfr[g].q[1] = *(const float4*)(Bsb + row * LDT + ks + 16 + lhi * 8);
      }
      #pragma unroll
      for (int g = 0; g < 4; ++g)
        #pragma unroll
        for (int mt = 0; mt < 2; ++mt)
          acc[g][mt] = __builtin_amdgcn_wmma_f32_16x16x32_f16(
              false, afr[mt].v, false, bfr[g].v, (short)0, acc[g][mt], false, false);
    }
    __syncthreads();       // everyone done reading tile t before its buffer is re-filled
  }

  // ---- fused gate epilogue: lane element r -> (m = mt*16 + lhi*8 + r, n = lm)
  const int n = bn + wn * 16 + lm;                  // 0..1023
  const float bi = b_ii[n] + b_hi[n];
  const float bf = b_if[n] + b_hf[n];
  const float bg = b_ig[n] + b_hg[n];
  const float bo = b_io[n] + b_ho[n];

  #pragma unroll
  for (int mt = 0; mt < 2; ++mt) {
    #pragma unroll
    for (int r = 0; r < 8; ++r) {
      int m = bm + wm * 32 + mt * 16 + lhi * 8 + r;
      float gi = acc[0][mt][r] + bi;
      float gf = acc[1][mt][r] + bf;
      float gg = acc[2][mt][r] + bg;
      float go = acc[3][mt][r] + bo;
      float iv = sigmoidf_(gi);
      float fv = sigmoidf_(gf);
      float gv = fast_tanhf_(gg);
      float ov = sigmoidf_(go);
      float pc = prev_c[(size_t)m * H_DIM + n];
      float cv = fv * pc + iv * gv;
      float hv = ov * fast_tanhf_(cv);
      out[(size_t)m * H_DIM + n] = hv;
      out[(size_t)B_DIM * H_DIM + (size_t)m * H_DIM + n] = cv;
    }
  }
}

// ---------------- host launch ----------------

extern "C" void kernel_launch(void* const* d_in, const int* in_sizes, int n_in,
                              void* d_out, int out_size, void* d_ws, size_t ws_size,
                              hipStream_t stream) {
  const float* input_ = (const float*)d_in[0];
  const float* prev_h = (const float*)d_in[1];
  const float* prev_c = (const float*)d_in[2];
  const float* W_ii = (const float*)d_in[3];  const float* b_ii = (const float*)d_in[4];
  const float* W_hi = (const float*)d_in[5];  const float* b_hi = (const float*)d_in[6];
  const float* W_if = (const float*)d_in[7];  const float* b_if = (const float*)d_in[8];
  const float* W_hf = (const float*)d_in[9];  const float* b_hf = (const float*)d_in[10];
  const float* W_ig = (const float*)d_in[11]; const float* b_ig = (const float*)d_in[12];
  const float* W_hg = (const float*)d_in[13]; const float* b_hg = (const float*)d_in[14];
  const float* W_io = (const float*)d_in[15]; const float* b_io = (const float*)d_in[16];
  const float* W_ho = (const float*)d_in[17]; const float* b_ho = (const float*)d_in[18];

  _Float16* X16 = (_Float16*)d_ws;                                      // 64 MB
  _Float16* W16 = (_Float16*)((char*)d_ws + (size_t)B_DIM * K_DIM * 2); // +16 MB

  {
    int total = B_DIM * (K_DIM / 4);
    pack_X_kernel<<<total / 256, 256, 0, stream>>>(input_, prev_h, X16);
  }
  {
    int total = 4 * H_DIM * (K_DIM / 4);
    pack_W_kernel<<<total / 256, 256, 0, stream>>>(W_ii, W_hi, W_if, W_hf,
                                                   W_ig, W_hg, W_io, W_ho, W16);
  }
  {
    dim3 grid(H_DIM / BN, B_DIM / BM);  // (32, 128)
    lstm_fused_kernel<<<grid, 256, 0, stream>>>(X16, W16, prev_c,
                                                b_ii, b_hi, b_if, b_hf,
                                                b_ig, b_hg, b_io, b_ho,
                                                (float*)d_out);
  }
}
